// MaskingModel_12970801234013
// MI455X (gfx1250) — matrix operand order repaired and checked by reading, not compiled, not run on hardware
//
#include <hip/hip_runtime.h>
#include <math.h>

// Problem constants (from reference setup)
#define BGR   256      // graphs
#define NNODE 64       // nodes per graph
#define MQ    8        // instruction vectors
#define DNN   2048     // dim_nodes
#define DQ    1024     // dim_questions
#define KTOP  16       // top-k

typedef __attribute__((ext_vector_type(16))) __bf16 v16bf;
typedef __attribute__((ext_vector_type(8)))  __bf16 v8bf;
typedef __attribute__((ext_vector_type(4)))  __bf16 v4bf;
typedef __attribute__((ext_vector_type(8)))  float  v8f;

union V16U { v16bf v; v8bf h[2]; };

static __device__ inline __bf16 tobf(float f) { return (__bf16)f; }
static __device__ inline v8f vzero8() { v8f z; for (int i = 0; i < 8; ++i) z[i] = 0.0f; return z; }

// Branch-free GELU (tanh form) -> keeps the writeback epilogue straight-line so
// the 64 accumulator VGPRs stay live in registers instead of spilling around
// libm erff's divergent control flow.
static __device__ inline float gelu_bf(float v) {
    float z = 0.7978845608028654f * v * (1.0f + 0.044715f * v * v);
    z = fminf(fmaxf(z, -30.0f), 30.0f);          // keep __expf finite, still branch-free
    float e = __expf(-2.0f * z);                 // v_exp_f32
    float th = (1.0f - e) / (1.0f + e);          // tanh(z)
    return 0.5f * v * (1.0f + th);
}

// LDS layout (element offsets in __bf16 units)
#define DENSE_STRIDE 1032
#define XA_STRIDE    40
#define WT_STRIDE    40
#define OFF_DENSE 0
#define OFF_INSTQ (NNODE * DENSE_STRIDE)            // 66048
#define OFF_XA    (OFF_INSTQ + 16 * DQ)             // 82432
#define OFF_WT    (OFF_XA + NNODE * XA_STRIDE)      // 84992
#define NBF       (OFF_WT + 256 * WT_STRIDE)        // 95232 bf16 elements
#define SMEM_BYTES (NBF * 2 + (MQ * NNODE) * 4 + 8 * 4 + 8 * 4 + NNODE * 4) // 192832 B

// ~193 KB dynamic LDS means exactly one 8-wave workgroup per WGP (2 waves/SIMD).
__global__ __launch_bounds__(256, 2)
void masking_fused(const float* __restrict__ x,
                   const float* __restrict__ u,
                   const float* __restrict__ W,
                   const float* __restrict__ bias,
                   float* __restrict__ out)
{
    extern __shared__ __align__(16) char smemraw[];
    __bf16* sm     = (__bf16*)smemraw;
    __bf16* dense  = sm + OFF_DENSE;   // [64][1032] bf16: GELU(X_b @ W + b)
    __bf16* instq  = sm + OFF_INSTQ;   // [16][1024] bf16: u_b rows 0..7, zeros 8..15
    __bf16* xa     = sm + OFF_XA;      // [64][40]   bf16 X tile (K-step 32)
    __bf16* wt     = sm + OFF_WT;      // [256][40]  bf16 W tile, stored transposed [n][k]
    float*  scoreS = (float*)(sm + NBF);
    float*  rowmax = scoreS + MQ * NNODE;
    float*  rowsum = rowmax + 8;
    float*  gateS  = rowsum + 8;

    const int t    = threadIdx.x;
    const int b    = blockIdx.x;
    const int w    = t >> 5;        // wave id (wave32)
    const int l    = t & 31;
    const int half = l >> 4;        // lane half selects K-subblock per ISA WMMA layout
    const int l15  = l & 15;

    const float* xg = x + (size_t)b * NNODE * DNN;

    // ---- Phase 0: stage instruction vectors u[:, b, :] to LDS as bf16 ----
    for (int i = 0; i < 8; ++i) {
        int e = (i * 256 + t) * 4;               // 8*1024 = 8192 elements
        int m = e >> 10, d = e & 1023;
        const float4 v = *(const float4*)(u + ((size_t)m * BGR + b) * DQ + d);
        v4bf p; p[0] = tobf(v.x); p[1] = tobf(v.y); p[2] = tobf(v.z); p[3] = tobf(v.w);
        *(v4bf*)(instq + m * DQ + d) = p;
    }
    { // zero pad rows 8..15 so the 16x16 score tile rows 8..15 are inert
        int row  = 8 + (t >> 5);
        int col0 = (t & 31) * 32;
        v8bf z; for (int q = 0; q < 8; ++q) z[q] = (__bf16)0.0f;
        for (int j = 0; j < 32; j += 8) *(v8bf*)(instq + row * DQ + col0 + j) = z;
    }

    // ---- Phase 1: dense = GELU(X_b[64,2048] @ W[2048,1024] + bias), bf16 WMMA ----
    const int wbase = w * 32;                    // wave's N strip within a 256-col chunk
    for (int nc = 0; nc < 4; ++nc) {
        const int ncol0 = nc * 256;
        v8f acc[4][2];
        for (int mt = 0; mt < 4; ++mt)
            for (int nt = 0; nt < 2; ++nt) acc[mt][nt] = vzero8();

        for (int kk = 0; kk < DNN; kk += 32) {
            __syncthreads();   // guard LDS tile overwrite (WAR)
            // X tile: 64 rows x 32 K, f32 -> bf16
            {
                int row = t >> 2, seg = t & 3;
                const float4 v0 = *(const float4*)(xg + (size_t)row * DNN + kk + seg * 8);
                const float4 v1 = *(const float4*)(xg + (size_t)row * DNN + kk + seg * 8 + 4);
                v8bf p;
                p[0] = tobf(v0.x); p[1] = tobf(v0.y); p[2] = tobf(v0.z); p[3] = tobf(v0.w);
                p[4] = tobf(v1.x); p[5] = tobf(v1.y); p[6] = tobf(v1.z); p[7] = tobf(v1.w);
                *(v8bf*)(xa + row * XA_STRIDE + seg * 8) = p;
            }
            // W tile: 32 K x 256 N, stored transposed wt[n][k] so B fragments are K-contiguous
            for (int i = 0; i < 8; ++i) {
                int e = (i * 256 + t) * 4;       // 32*256 = 8192 elements
                int k = e >> 8, n4 = e & 255;
                const float4 v = *(const float4*)(W + (size_t)(kk + k) * DQ + ncol0 + n4);
                wt[(n4 + 0) * WT_STRIDE + k] = tobf(v.x);
                wt[(n4 + 1) * WT_STRIDE + k] = tobf(v.y);
                wt[(n4 + 2) * WT_STRIDE + k] = tobf(v.z);
                wt[(n4 + 3) * WT_STRIDE + k] = tobf(v.w);
            }
            __syncthreads();

            // B fragments: lane l<16 holds K=0..15 of column n; l>=16 holds K=16..31
            V16U bf[2];
            const int kb = half ? 16 : 0;
            for (int nt = 0; nt < 2; ++nt) {
                const int col = wbase + nt * 16 + l15;
                bf[nt].h[0] = *(const v8bf*)(wt + col * WT_STRIDE + kb);
                bf[nt].h[1] = *(const v8bf*)(wt + col * WT_STRIDE + kb + 8);
            }
            // A fragments: lane l<16 holds K=0..15 of row m; l>=16 holds K=16..31
            const int k0 = half ? 16 : 0;
            for (int mt = 0; mt < 4; ++mt) {
                V16U af;
                const int row = mt * 16 + l15;
                af.h[0] = *(const v8bf*)(xa + row * XA_STRIDE + k0);
                af.h[1] = *(const v8bf*)(xa + row * XA_STRIDE + k0 + 8);
                for (int nt = 0; nt < 2; ++nt)
                    acc[mt][nt] = __builtin_amdgcn_wmma_f32_16x16x32_bf16(
                        false, af.v, false, bf[nt].v, (short)0, acc[mt][nt], false, false);
            }
        }

        // writeback: bias + branch-free GELU, f32 -> bf16 into dense
        for (int mt = 0; mt < 4; ++mt)
            for (int nt = 0; nt < 2; ++nt) {
                const int colg = ncol0 + wbase + nt * 16 + l15;
                const float bv = bias[colg];
                for (int j = 0; j < 8; ++j) {
                    const int rowg = mt * 16 + j + half * 8;  // C/D layout: VGPR j -> M=j (+8 for hi lanes)
                    dense[rowg * DENSE_STRIDE + colg] = tobf(gelu_bf(acc[mt][nt][j] + bv));
                }
            }
    }
    __syncthreads();

    // ---- Phase 2: scores[8,64] = u_b @ dense^T via WMMA (waves 0..3, one node tile each) ----
    if (w < 4) {
        v8f sacc = vzero8();
        const int node0 = w * 16;
        const int k0 = half ? 16 : 0;
        const int kb = half ? 16 : 0;
        const int node = node0 + l15;
        for (int kk = 0; kk < DQ; kk += 32) {
            V16U af, bf;
            af.h[0] = *(const v8bf*)(instq + l15 * DQ + kk + k0);
            af.h[1] = *(const v8bf*)(instq + l15 * DQ + kk + k0 + 8);
            bf.h[0] = *(const v8bf*)(dense + node * DENSE_STRIDE + kk + kb);
            bf.h[1] = *(const v8bf*)(dense + node * DENSE_STRIDE + kk + kb + 8);
            sacc = __builtin_amdgcn_wmma_f32_16x16x32_bf16(
                false, af.v, false, bf.v, (short)0, sacc, false, false);
        }
        if (half == 0) {  // rows m=0..7 live in low lanes; hi lanes carry m=8..15 (zero pad)
            for (int j = 0; j < MQ; ++j)
                scoreS[j * NNODE + node0 + l15] = sacc[j];
        }
    }
    __syncthreads();

    // ---- Phase 3: softmax over nodes, sum over m, hard top-16 mask ----
    if (t < MQ) {
        float mx = -3.0e38f;
        for (int n = 0; n < NNODE; ++n) mx = fmaxf(mx, scoreS[t * NNODE + n]);
        float s = 0.0f;
        for (int n = 0; n < NNODE; ++n) s += __expf(scoreS[t * NNODE + n] - mx);
        rowmax[t] = mx; rowsum[t] = s;
    }
    __syncthreads();
    if (t < NNODE) {
        float g = 0.0f;
        for (int m = 0; m < MQ; ++m)
            g += __expf(scoreS[m * NNODE + t] - rowmax[m]) / rowsum[m];
        gateS[t] = g;
    }
    __syncthreads();
    if (t < NNODE) {
        const float g = gateS[t];
        int cnt = 0;
        for (int j = 0; j < NNODE; ++j) cnt += (gateS[j] > g) ? 1 : 0;
        // gate[n] >= kth-largest  <=>  fewer than K strictly-greater values
        out[(size_t)b * NNODE + t] = (cnt < KTOP) ? 1.0f : 0.0f;
    }
}

extern "C" void kernel_launch(void* const* d_in, const int* in_sizes, int n_in,
                              void* d_out, int out_size, void* d_ws, size_t ws_size,
                              hipStream_t stream) {
    (void)in_sizes; (void)n_in; (void)out_size; (void)d_ws; (void)ws_size;
    const float* x    = (const float*)d_in[0];  // [16384, 2048]
    const float* u    = (const float*)d_in[1];  // [8, 256, 1024]
    // d_in[2] = batch (implied by structure), d_in[3] = edge_index (unused), d_in[6] = size
    const float* W    = (const float*)d_in[4];  // [2048, 1024]
    const float* bias = (const float*)d_in[5];  // [1024]
    float* out = (float*)d_out;                 // [16384]

    hipFuncSetAttribute((const void*)masking_fused,
                        hipFuncAttributeMaxDynamicSharedMemorySize, SMEM_BYTES);
    masking_fused<<<BGR, 256, SMEM_BYTES, stream>>>(x, u, W, bias, out);
}
